// LocalAttentionFusion_69939247448723
// MI455X (gfx1250) — matrix operand order repaired and checked by reading, not compiled, not run on hardware
//
#include <hip/hip_runtime.h>
#include <hip/hip_bf16.h>
#include <cstdint>
#include <cstddef>

typedef __attribute__((ext_vector_type(16))) _Float16 v16h;
typedef __attribute__((ext_vector_type(8)))  float    v8f;

#define ROI 7
#define CH  256

// ---------------- WMMA fragment helpers ----------------
union FragU { uint4 q[2]; v16h v; };

// A fragment (16x32 f16): base pre-biased by lhi*16; 16 contiguous halves
__device__ __forceinline__ v16h load_a_frag(const _Float16* p) {
  FragU u;
  u.q[0] = *(const uint4*)(p);
  u.q[1] = *(const uint4*)(p + 8);
  return u.v;
}
// B fragment (32x16 f16): base pre-biased by lhi*16; 16 contiguous halves
__device__ __forceinline__ v16h load_b_frag(const _Float16* p) {
  FragU u;
  u.q[0] = *(const uint4*)(p);
  u.q[1] = *(const uint4*)(p + 8);
  return u.v;
}
__device__ __forceinline__ v8f wmma_f16(v16h a, v16h b, v8f c) {
  return __builtin_amdgcn_wmma_f32_16x16x32_f16(false, a, false, b, (short)0, c,
                                                false, false);
}

// ---------------- LDS layout (bytes); 2 samples per workgroup ----------------
constexpr int XP1_STRIDE = 520;   // halves (512 ch + 8 pad -> 260 dw, conflict-free)
constexpr int XP2_STRIDE = 264;   // halves (256 ch + 8 pad -> 132 dw)
constexpr int HIN_STRIDE = 264;   // halves
constexpr int G_STRIDE   = 68;    // floats

constexpr int XP1_SAMP_H = 81 * XP1_STRIDE;        // 42120 halves / sample
constexpr int XP2_SAMP_H = 81 * XP2_STRIDE;        // 21384 halves / sample
constexpr int HIN_SAMP_H = 17408;                  // halves / sample (padded)
constexpr int G_SAMP_F   = 4352;                   // floats / sample

constexpr int OFF_XP1 = 0;                         // 2 * 84240 = 168480
constexpr int OFF_HIN = 0;                         // alias in XP1 region
constexpr int OFF_G   = 84240;                     // alias in XP1b region
constexpr int OFF_XP2 = 168480;                    // 2 * 42768 -> ends 254016
constexpr int OFF_AB  = 254016;                    // 1024 f32
constexpr int OFF_WGB = OFF_AB + 4096;             // 64 f32
constexpr int OFF_W2W = OFF_WGB + 256;             // 128 f32
constexpr int OFF_W2B = OFF_W2W + 512;             // 2 f32 (+pad)
constexpr int OFF_TAB = OFF_W2B + 16;              // 2 samples * 176 B
constexpr int OFF_W0  = OFF_TAB + 352;             // 98 f32
constexpr int SMEM_BYTES = OFF_W0 + 400;           // ~259.7 KB (< 320 KB/WGP)

// ---------------- prep: weights -> f16 [tap][co][ci], fold BN ----------------
__global__ void __launch_bounds__(256) prep_kernel(
    const float* __restrict__ c1w, const float* __restrict__ c1b,
    const float* __restrict__ s1, const float* __restrict__ b1,
    const float* __restrict__ m1, const float* __restrict__ v1,
    const float* __restrict__ c2w, const float* __restrict__ c2b,
    const float* __restrict__ s2, const float* __restrict__ b2,
    const float* __restrict__ m2, const float* __restrict__ v2,
    const float* __restrict__ g1w,
    _Float16* __restrict__ w1, _Float16* __restrict__ w2,
    _Float16* __restrict__ wg, float* __restrict__ ab)
{
  const int tid = blockIdx.x * blockDim.x + threadIdx.x;
  const int stride = gridDim.x * blockDim.x;
  for (int i = tid; i < 9 * 256 * 512; i += stride) {
    int ci = i & 511;
    int r  = i >> 9;
    int co = r & 255;
    int tap = r >> 8;
    w1[i] = (_Float16)c1w[(co * 512 + ci) * 9 + tap];
  }
  for (int i = tid; i < 9 * 256 * 256; i += stride) {
    int ci = i & 255;
    int r  = i >> 8;
    int co = r & 255;
    int tap = r >> 8;
    w2[i] = (_Float16)c2w[(co * 256 + ci) * 9 + tap];
  }
  for (int i = tid; i < 64 * 256; i += stride) wg[i] = (_Float16)g1w[i];
  for (int i = tid; i < 256; i += stride) {
    float inv1 = s1[i] * rsqrtf(v1[i] + 1e-5f);
    ab[i]       = inv1;
    ab[256 + i] = b1[i] + (c1b[i] - m1[i]) * inv1;
    float inv2 = s2[i] * rsqrtf(v2[i] + 1e-5f);
    ab[512 + i] = inv2;
    ab[768 + i] = b2[i] + (c2b[i] - m2[i]) * inv2;
  }
}

// ---- 3x3 implicit-GEMM conv core: 2 Mtiles x 7 Ntiles (N spans 2 samples) ----
// K loop fully unrolled: all loads use constant immediate offsets off fixed
// per-lane base pointers -> no address VALU inside the MMA stream, no WAR NOPs.
template <int KSTEPS>
__device__ __forceinline__ void conv3x3_wmma2(
    const _Float16* __restrict__ wts,   // [9][256][K] f16
    const _Float16* inLds, int inStride, int sampHalves,
    int wv, int lane, v8f (&acc)[2][7])
{
  const int l15 = lane & 15, lhi = lane >> 4;
  constexpr int Kch = KSTEPS * 32;
  v8f zro = {};
#pragma unroll
  for (int mi = 0; mi < 2; ++mi)
#pragma unroll
    for (int nt = 0; nt < 7; ++nt) acc[mi][nt] = zro;

  int yy[7], xx[7], sOff[7], val[7];
#pragma unroll
  for (int nt = 0; nt < 7; ++nt) {
    int n = nt * 16 + l15;            // 0..111 ; [0,49) samp a, [49,98) samp b
    int samp = (n >= 49) ? 1 : 0;
    int np = n - samp * 49;
    val[nt]  = (n < 98) ? 1 : 0;
    sOff[nt] = samp * sampHalves;
    yy[nt] = np / 7;
    xx[nt] = np - yy[nt] * 7;
  }

#pragma unroll 1
  for (int tap = 0; tap < 9; ++tap) {
    int dy = tap / 3, dx = tap - dy * 3;
    // per-tap fixed base pointers (include all per-lane components)
    const _Float16* bp[7];
#pragma unroll
    for (int nt = 0; nt < 7; ++nt) {
      int off = val[nt] ? (sOff[nt] + ((yy[nt] + dy) * 9 + (xx[nt] + dx)) * inStride)
                        : 0;            // invalid -> zero border row of sample a
      bp[nt] = inLds + off + lhi * 16;
    }
    const _Float16* wrowA = wts + (size_t)(tap * 256 + wv * 32 + l15) * Kch + lhi * 16;
    const _Float16* wrowB = wrowA + (size_t)16 * Kch;
#pragma unroll
    for (int ks = 0; ks < KSTEPS; ++ks) {
      v16h a0 = load_a_frag(wrowA + ks * 32);
      v16h a1 = load_a_frag(wrowB + ks * 32);
#pragma unroll
      for (int nt = 0; nt < 7; ++nt) {
        v16h bf = load_b_frag(bp[nt] + ks * 32);
        acc[0][nt] = wmma_f16(a0, bf, acc[0][nt]);
        acc[1][nt] = wmma_f16(a1, bf, acc[1][nt]);
      }
    }
  }
}

// ---------------- fused pipeline: 2 samples per workgroup ----------------
__global__ void __launch_bounds__(256) fused_kernel(
    const float* __restrict__ frgb, const float* __restrict__ ftir,
    const int* __restrict__ coords,
    const _Float16* __restrict__ w1, const _Float16* __restrict__ w2,
    const _Float16* __restrict__ wg, const float* __restrict__ ab,
    const float* __restrict__ wg1b, const float* __restrict__ wg2w,
    const float* __restrict__ wg2b, float* __restrict__ out)
{
  __shared__ __align__(16) unsigned char smem[SMEM_BYTES];
  _Float16* xp1  = (_Float16*)(smem + OFF_XP1);
  _Float16* xp2  = (_Float16*)(smem + OFF_XP2);
  _Float16* hin  = (_Float16*)(smem + OFF_HIN);
  float*    gbuf = (float*)(smem + OFF_G);
  float*    abL  = (float*)(smem + OFF_AB);
  float*    gbL  = (float*)(smem + OFF_WGB);
  float*    w2wL = (float*)(smem + OFF_W2W);
  float*    w2bL = (float*)(smem + OFF_W2B);
  float*    w0L  = (float*)(smem + OFF_W0);

  const int tid  = threadIdx.x;
  const int s0   = blockIdx.x * 2;      // samples s0, s0+1 (same b: 128 is even)
  const int b    = s0 >> 7;
  const int lane = tid & 31;
  const int wv   = tid >> 5;
  const int l15  = lane & 15;
  const int lhi  = lane >> 4;

  // ---- init: bilinear tables for both samples (reference axis_idx) ----
  if (tid < 28) {
    int samp = tid / 14, r = tid - samp * 14;
    int axis = r / 7, j = r - axis * 7;        // axis 0 = y(H), 1 = x(W)
    int c = coords[(s0 + samp) * 2 + (axis == 0 ? 1 : 0)];
    int start = max(0, c - 3);
    int end   = min(64, c + 4);
    int L = end - start;
    float src = (j + 0.5f) * (float)L * (1.0f / 7.0f) - 0.5f;
    src = fminf(fmaxf(src, 0.0f), (float)(L - 1));
    int i0 = (int)src;
    float wfrac = src - (float)i0;
    int i1 = min(i0 + 1, L - 1);
    int*   tI = (int*)(smem + OFF_TAB + samp * 176);
    float* tF = (float*)(smem + OFF_TAB + samp * 176 + 112);
    tI[axis * 14 + j]     = start + i0;
    tI[axis * 14 + 7 + j] = start + i1;
    tF[axis * 7 + j]      = wfrac;
  }
  for (int i = tid; i < 1024; i += 256) abL[i] = ab[i];
  if (tid < 64) gbL[tid] = wg1b[tid];
  if (tid < 128) w2wL[tid] = wg2w[tid];
  if (tid < 2) w2bL[tid] = wg2b[tid];
  {
    uint32_t* z1 = (uint32_t*)xp1;
    for (int i = tid; i < 2 * 81 * XP1_STRIDE / 2; i += 256) z1[i] = 0u;
    uint32_t* z2 = (uint32_t*)xp2;
    for (int i = tid; i < 2 * 81 * XP2_STRIDE / 2; i += 256) z2[i] = 0u;
  }
  __syncthreads();

  // ---- ROI bilinear extract (f32 math, f16 store) into XP1 interiors ----
  {
    const float* baseR = frgb + ((size_t)b * CH << 12);
    const float* baseT = ftir + ((size_t)b * CH << 12);
    for (int idx = tid; idx < 2 * 512 * 49; idx += 256) {
      int samp = (idx >= 25088) ? 1 : 0;
      int k = idx - samp * 25088;
      int ci = k / 49;
      int n  = k - ci * 49;
      int i  = n / 7, j = n - i * 7;
      const int*   tI = (const int*)(smem + OFF_TAB + samp * 176);
      const float* tF = (const float*)(smem + OFF_TAB + samp * 176 + 112);
      int Y0 = tI[i], Y1 = tI[7 + i], X0 = tI[14 + j], X1 = tI[21 + j];
      float fy1 = tF[i], fx1 = tF[7 + j];
      float fy0 = 1.0f - fy1, fx0 = 1.0f - fx1;
      const float* f = (ci < 256 ? baseR : baseT) + ((size_t)(ci & 255) << 12);
      float v = fy0 * (fx0 * f[Y0 * 64 + X0] + fx1 * f[Y0 * 64 + X1]) +
                fy1 * (fx0 * f[Y1 * 64 + X0] + fx1 * f[Y1 * 64 + X1]);
      xp1[samp * XP1_SAMP_H + ((i + 1) * 9 + (j + 1)) * XP1_STRIDE + ci] = (_Float16)v;
    }
  }
  __syncthreads();

  // ---- conv1 (512 -> 256) + BN1 + ReLU -> XP2 ----
  {
    v8f acc[2][7];
    conv3x3_wmma2<16>(w1, xp1, XP1_STRIDE, XP1_SAMP_H, wv, lane, acc);
#pragma unroll
    for (int mi = 0; mi < 2; ++mi)
#pragma unroll
      for (int nt = 0; nt < 7; ++nt) {
        int n = nt * 16 + l15;
        int samp = (n >= 49) ? 1 : 0;
        int np = n - samp * 49;
        if (n < 98) {
          int y = np / 7, x = np - y * 7;
          _Float16* dst = xp2 + samp * XP2_SAMP_H + ((y + 1) * 9 + (x + 1)) * XP2_STRIDE;
#pragma unroll
          for (int r = 0; r < 8; ++r) {
            int co = (wv * 2 + mi) * 16 + r + lhi * 8;
            float v = acc[mi][nt][r] * abL[co] + abL[256 + co];
            dst[co] = (_Float16)fmaxf(v, 0.0f);
          }
        }
      }
  }
  __syncthreads();

  // ---- conv2 (256 -> 256) + BN2 + ReLU -> HIN[n][co] (XP1 now dead) ----
#pragma unroll
  for (int samp = 0; samp < 2; ++samp) {   // zero tail rows 49..63 of each HIN
    uint32_t* z = (uint32_t*)(hin + samp * HIN_SAMP_H + 49 * HIN_STRIDE);
    for (int i = tid; i < 15 * HIN_STRIDE / 2; i += 256) z[i] = 0u;
  }
  {
    v8f acc[2][7];
    conv3x3_wmma2<8>(w2, xp2, XP2_STRIDE, XP2_SAMP_H, wv, lane, acc);
#pragma unroll
    for (int mi = 0; mi < 2; ++mi)
#pragma unroll
      for (int nt = 0; nt < 7; ++nt) {
        int n = nt * 16 + l15;
        int samp = (n >= 49) ? 1 : 0;
        int np = n - samp * 49;
        if (n < 98) {
          _Float16* dst = hin + samp * HIN_SAMP_H + np * HIN_STRIDE;
#pragma unroll
          for (int r = 0; r < 8; ++r) {
            int co = (wv * 2 + mi) * 16 + r + lhi * 8;
            float v = acc[mi][nt][r] * abL[512 + co] + abL[768 + co];
            dst[co] = (_Float16)fmaxf(v, 0.0f);
          }
        }
      }
  }
  __syncthreads();

  // ---- wg1 (1x1, 256 -> 64) + bias + ReLU -> G[n][h] f32 (28 tiles / 8 waves) --
  {
    v8f accW[4];
    v8f zro = {};
    int mtW[4], npW[4], sampW[4], okW[4], nW[4];
    const _Float16* aBase[4];
    const _Float16* bBase[4];
#pragma unroll
    for (int u = 0; u < 4; ++u) {
      int t = wv + 8 * u;
      okW[u] = (t < 28) ? 1 : 0;
      int tt = okW[u] ? t : 0;
      int mt = tt / 7, nt = tt - mt * 7;
      mtW[u] = mt;
      int n = nt * 16 + l15;
      int samp = (n >= 49) ? 1 : 0;
      int np = n - samp * 49;
      nW[u] = n; npW[u] = np; sampW[u] = samp;
      aBase[u] = wg + (size_t)(mt * 16 + l15) * 256 + lhi * 16;
      bBase[u] = hin + samp * HIN_SAMP_H + np * HIN_STRIDE + lhi * 16;
      accW[u] = zro;
    }
#pragma unroll
    for (int ks = 0; ks < 8; ++ks) {
#pragma unroll
      for (int u = 0; u < 4; ++u) {
        if (okW[u]) {
          v16h a  = load_a_frag(aBase[u] + ks * 32);
          v16h bb = load_b_frag(bBase[u] + ks * 32);
          accW[u] = wmma_f16(a, bb, accW[u]);
        }
      }
    }
#pragma unroll
    for (int u = 0; u < 4; ++u) {
      if (okW[u] && nW[u] < 98) {
        float* g = gbuf + sampW[u] * G_SAMP_F + npW[u] * G_STRIDE;
#pragma unroll
        for (int r = 0; r < 8; ++r) {
          int co = mtW[u] * 16 + r + lhi * 8;
          g[co] = fmaxf(accW[u][r] + gbL[co], 0.0f);
        }
      }
    }
  }
  __syncthreads();

  // ---- wg2 (1x1, 64 -> 2) + 2-way softmax ----
  if (tid < 98) {
    int samp = tid / 49, n = tid - samp * 49;
    float l0 = w2bL[0], l1 = w2bL[1];
    const float* g = gbuf + samp * G_SAMP_F + n * G_STRIDE;
#pragma unroll
    for (int h = 0; h < 64; ++h) {
      float gv = g[h];
      l0 += gv * w2wL[h];
      l1 += gv * w2wL[64 + h];
    }
    w0L[tid] = 1.0f / (1.0f + __expf(l1 - l0));
  }
  __syncthreads();

  // ---- fuse: f32 bilinear recompute (L2-resident feats) * softmax weights ----
  {
    const float* baseR = frgb + ((size_t)b * CH << 12);
    const float* baseT = ftir + ((size_t)b * CH << 12);
    for (int idx = tid; idx < 2 * CH * 49; idx += 256) {
      int samp = (idx >= CH * 49) ? 1 : 0;
      int k = idx - samp * CH * 49;
      int c = k / 49;
      int n = k - c * 49;
      int i = n / 7, j = n - i * 7;
      const int*   tI = (const int*)(smem + OFF_TAB + samp * 176);
      const float* tF = (const float*)(smem + OFF_TAB + samp * 176 + 112);
      int Y0 = tI[i], Y1 = tI[7 + i], X0 = tI[14 + j], X1 = tI[21 + j];
      float fy1 = tF[i], fx1 = tF[7 + j];
      float fy0 = 1.0f - fy1, fx0 = 1.0f - fx1;
      const float* fr = baseR + ((size_t)c << 12);
      const float* ft = baseT + ((size_t)c << 12);
      float rgbv = fy0 * (fx0 * fr[Y0 * 64 + X0] + fx1 * fr[Y0 * 64 + X1]) +
                   fy1 * (fx0 * fr[Y1 * 64 + X0] + fx1 * fr[Y1 * 64 + X1]);
      float tirv = fy0 * (fx0 * ft[Y0 * 64 + X0] + fx1 * ft[Y0 * 64 + X1]) +
                   fy1 * (fx0 * ft[Y1 * 64 + X0] + fx1 * ft[Y1 * 64 + X1]);
      float wv0 = w0L[samp * 49 + n];
      out[((size_t)(s0 + samp) * CH + c) * 49 + n] = wv0 * rgbv + (1.0f - wv0) * tirv;
    }
  }
}

extern "C" void kernel_launch(void* const* d_in, const int* in_sizes, int n_in,
                              void* d_out, int out_size, void* d_ws, size_t ws_size,
                              hipStream_t stream) {
  const float* feat_rgb = (const float*)d_in[0];
  const float* feat_tir = (const float*)d_in[1];
  const int*   coords   = (const int*)d_in[2];
  const float* c1w = (const float*)d_in[3];
  const float* c1b = (const float*)d_in[4];
  const float* s1  = (const float*)d_in[5];
  const float* b1  = (const float*)d_in[6];
  const float* m1  = (const float*)d_in[7];
  const float* v1  = (const float*)d_in[8];
  const float* c2w = (const float*)d_in[9];
  const float* c2b = (const float*)d_in[10];
  const float* s2  = (const float*)d_in[11];
  const float* b2  = (const float*)d_in[12];
  const float* m2  = (const float*)d_in[13];
  const float* v2  = (const float*)d_in[14];
  const float* g1w = (const float*)d_in[15];
  const float* g1b = (const float*)d_in[16];
  const float* g2w = (const float*)d_in[17];
  const float* g2b = (const float*)d_in[18];
  (void)in_sizes; (void)n_in; (void)out_size; (void)ws_size;

  _Float16* w1f = (_Float16*)d_ws;                 // 9*256*512 halves
  _Float16* w2f = w1f + 9 * 256 * 512;             // 9*256*256 halves
  _Float16* wgf = w2f + 9 * 256 * 256;             // 64*256 halves
  float*    ab  = (float*)(wgf + 64 * 256);        // 1024 f32

  prep_kernel<<<1024, 256, 0, stream>>>(c1w, c1b, s1, b1, m1, v1,
                                        c2w, c2b, s2, b2, m2, v2,
                                        g1w, w1f, w2f, wgf, ab);
  fused_kernel<<<512, 256, 0, stream>>>(feat_rgb, feat_tir, coords,
                                        w1f, w2f, wgf, ab, g1b, g2w, g2b,
                                        (float*)d_out);
}
